// FeastLayer_78881369358600
// MI455X (gfx1250) — compile-verified, matrix operand order checked by reading, not run on hardware
//
#include <hip/hip_runtime.h>
#include <hip/hip_bf16.h>

// ---------------------------------------------------------------------------
// FEAST graph conv for MI455X (gfx1250, wave32).
//  B=4, N=20000, K=16 neighbors, M=9 kernels, C=64 in, O=64 out.
//  HBM-gather bound (~350MB of neighbor traffic); the s*W GEMM (5.9 GFLOP)
//  is pushed through v_wmma_f32_16x16x32_bf16 so it costs nothing.
// ---------------------------------------------------------------------------

typedef __attribute__((ext_vector_type(16))) __bf16         v16bf;
typedef __attribute__((ext_vector_type(8)))  float          v8f;
typedef __attribute__((ext_vector_type(8)))  unsigned short ushort8;

__device__ __forceinline__ unsigned short f2bf(float f) {
    // round-to-nearest-even f32 -> bf16 bits
    unsigned int ui = __float_as_uint(f);
    unsigned int r  = ui + 0x7FFFu + ((ui >> 16) & 1u);
    return (unsigned short)(r >> 16);
}

// ---------------------------------------------------------------------------
// Kernel 1: per-node projections.
//   uxc[bn,m] = dot(x[bn,:], u[m,:]) + c[m]   (fold +c of the logits here)
//   vxv[bn,m] = dot(x[bn,:], v[m,:])
// ---------------------------------------------------------------------------
__global__ void feast_pre(const float* __restrict__ x,
                          const float* __restrict__ u,
                          const float* __restrict__ v,
                          const float* __restrict__ cvec,
                          float* __restrict__ uxc,
                          float* __restrict__ vxv,
                          int totalBN) {
    int tid = blockIdx.x * blockDim.x + threadIdx.x;
    if (tid >= totalBN * 9) return;
    int bn = tid / 9;
    int m  = tid - bn * 9;
    const float4* xr = (const float4*)(x + (long long)bn * 64);
    const float4* ur = (const float4*)(u + m * 64);
    const float4* vr = (const float4*)(v + m * 64);
    float du = 0.f, dv = 0.f;
#pragma unroll
    for (int i = 0; i < 16; ++i) {
        float4 xv = xr[i], uv = ur[i], vv = vr[i];
        du += xv.x * uv.x + xv.y * uv.y + xv.z * uv.z + xv.w * uv.w;
        dv += xv.x * vv.x + xv.y * vv.y + xv.z * vv.z + xv.w * vv.w;
    }
    uxc[tid] = du + cvec[m];
    vxv[tid] = dv;
}

// ---------------------------------------------------------------------------
// Kernel 2: pack W[M,O,C] -> bf16 WMMA-B fragments W'[mc = m*64+c, o].
// Layout: wfrag[((t*18 + kc)*32 + lane)*16 + i], t = O tile (o = t*16+(lane&15)),
// kc = 32-wide K chunk. Element i / lane-half map to K exactly like the ISA
// 16-bit A-matrix table (mirrored for B):
//   K = kc*32 + (i&7) + (i&8 ? 16 : 0) + (lane>=16 ? 8 : 0)
// ---------------------------------------------------------------------------
__global__ void feast_packW(const float* __restrict__ W,
                            unsigned short* __restrict__ wfrag) {
    int tid = blockIdx.x * blockDim.x + threadIdx.x;
    if (tid >= 4 * 18 * 32 * 16) return;
    int i    = tid & 15;
    int lane = (tid >> 4) & 31;
    int rest = tid >> 9;          // t*18 + kc
    int kc   = rest % 18;
    int t    = rest / 18;
    int half = lane >> 4;
    int K    = kc * 32 + (i & 7) + ((i & 8) ? 16 : 0) + (half ? 8 : 0);
    int m    = K >> 6;            // K = m*64 + c
    int cc   = K & 63;
    int o    = t * 16 + (lane & 15);
    wfrag[tid] = f2bf(W[m * 4096 + o * 64 + cc]);
}

// ---------------------------------------------------------------------------
// Kernel 3: main. One wave32 per 16-node tile of one batch.
//   1) load adj tile, softmax over M per (node,k) edge (q=0 for padding),
//   2) s[node,m,c] = inv_deg * sum_k q * x[neighbor,c]  (single gather of each
//      neighbor row, float2 per lane, coalesced), bf16 into LDS,
//   3) out[16x64] = s[16x576] * W'[576x64] via v_wmma_f32_16x16x32_bf16, + bias.
// ---------------------------------------------------------------------------
__global__ __launch_bounds__(32) void feast_main(
        const float* __restrict__ x,
        const int*   __restrict__ adj,
        const float* __restrict__ bias,
        const float* __restrict__ uxc,
        const float* __restrict__ vxv,
        const unsigned short* __restrict__ wfrag,
        float* __restrict__ out,
        int N) {
    const int lane          = threadIdx.x;
    const int tilesPerBatch = (N + 15) >> 4;
    const int b             = blockIdx.x / tilesPerBatch;
    const int n0            = (blockIdx.x % tilesPerBatch) << 4;

    // s rows padded 576 -> 584 (stride 1168B = 73*16B) so every A-fragment
    // ds_load_b128 stays 16B aligned and bank patterns rotate.
    __shared__ __align__(16) unsigned short s_lds[16][584];
    __shared__ float q_lds[16][16][10];
    __shared__ int   adj_lds[16][16];

    // ---- load adjacency tile (256 ints, 8 per lane) ----
    for (int p = lane; p < 256; p += 32) {
        int node = p >> 4, k = p & 15;
        int ng   = n0 + node;
        adj_lds[node][k] = (ng < N) ? adj[((long long)b * N + ng) * 16 + k] : 0;
    }
    __syncthreads();

    // ---- softmax over M per edge ----
    for (int p = lane; p < 256; p += 32) {
        int node = p >> 4, k = p & 15;
        int j = adj_lds[node][k];
        if (j > 0) {
            const float* up = uxc + ((long long)b * N + (n0 + node)) * 9;
            const float* vp = vxv + ((long long)b * N + (j - 1)) * 9;
            float lg[9];
            float mx = -1e30f;
#pragma unroll
            for (int m = 0; m < 9; ++m) { lg[m] = up[m] + vp[m]; mx = fmaxf(mx, lg[m]); }
            float s = 0.f;
#pragma unroll
            for (int m = 0; m < 9; ++m) { lg[m] = __expf(lg[m] - mx); s += lg[m]; }
            float inv = 1.0f / s;
#pragma unroll
            for (int m = 0; m < 9; ++m) q_lds[node][k][m] = lg[m] * inv;
        } else {
            // padded neighbor: gathered feature is zero, so q is irrelevant -> 0
#pragma unroll
            for (int m = 0; m < 9; ++m) q_lds[node][k][m] = 0.f;
        }
    }
    __syncthreads();

    // ---- s = inv_deg * sum_k q * x_j  (lanes own 2 feature columns) ----
    const int c0 = lane * 2;
    for (int node = 0; node < 16; ++node) {
        int deg = 0;
#pragma unroll
        for (int k = 0; k < 16; ++k) deg += (adj_lds[node][k] > 0);
        float inv_deg = deg > 0 ? 1.0f / (float)deg : 0.0f;
        float acc[9][2] = {};
        for (int k = 0; k < 16; ++k) {
            int j = adj_lds[node][k];
            if (j <= 0) continue;
            float2 xv = *(const float2*)(x + ((long long)b * N + (j - 1)) * 64 + c0);
#pragma unroll
            for (int m = 0; m < 9; ++m) {
                float qv = q_lds[node][k][m];        // LDS broadcast
                acc[m][0] += qv * xv.x;
                acc[m][1] += qv * xv.y;
            }
        }
#pragma unroll
        for (int m = 0; m < 9; ++m) {
            s_lds[node][m * 64 + c0]     = f2bf(acc[m][0] * inv_deg);
            s_lds[node][m * 64 + c0 + 1] = f2bf(acc[m][1] * inv_deg);
        }
    }
    __syncthreads();

    // ---- WMMA: D[16 nodes x 64 out] = A(s, 16x576) * B(W', 576x64) ----
    const int row  = lane & 15;   // A row (node) / B,D column-in-tile
    const int half = lane >> 4;
#pragma unroll
    for (int t = 0; t < 4; ++t) {
        v8f acc = {0.f, 0.f, 0.f, 0.f, 0.f, 0.f, 0.f, 0.f};
        for (int kc = 0; kc < 18; ++kc) {
            // A fragment: two 16B LDS loads cover i=0..7 (K base) and i=8..15 (K+16)
            union { ushort8 h[2]; v16bf v; } A;
            A.h[0] = *(const ushort8*)&s_lds[row][kc * 32 + half * 8];
            A.h[1] = *(const ushort8*)&s_lds[row][kc * 32 + 16 + half * 8];
            // B fragment: 32B contiguous (two global_load_b128)
            v16bf Bf = *(const v16bf*)(wfrag + (((t * 18 + kc) * 32 + lane) << 4));
            acc = __builtin_amdgcn_wmma_f32_16x16x32_bf16(
                false, A.v, false, Bf, (short)0, acc, false, false);
        }
        // C/D layout: VGPR r, lane l -> (M = r + 8*half [node], N = l&15 [o in tile])
        int o   = t * 16 + row;
        float bo = bias[o];
#pragma unroll
        for (int r = 0; r < 8; ++r) {
            int node = r + half * 8;
            int ng   = n0 + node;
            if (ng < N)
                out[((long long)b * N + ng) * 64 + o] = acc[r] + bo;
        }
    }
}

// ---------------------------------------------------------------------------
// Launch. Inputs (setup_inputs order): x, adj, W, b, u, v, c.
// Workspace: uxc[B*N*9] f32 | vxv[B*N*9] f32 | wfrag[36864] bf16  (~5.8 MB)
// ---------------------------------------------------------------------------
extern "C" void kernel_launch(void* const* d_in, const int* in_sizes, int n_in,
                              void* d_out, int out_size, void* d_ws, size_t ws_size,
                              hipStream_t stream) {
    const float* x    = (const float*)d_in[0];
    const int*   adj  = (const int*)d_in[1];
    const float* W    = (const float*)d_in[2];
    const float* bias = (const float*)d_in[3];
    const float* u    = (const float*)d_in[4];
    const float* v    = (const float*)d_in[5];
    const float* cvec = (const float*)d_in[6];

    const int B = 4, C = 64;
    const int N  = in_sizes[0] / (B * C);
    const long long BN = (long long)B * N;

    float*          uxc   = (float*)d_ws;
    float*          vxv   = uxc + BN * 9;
    unsigned short* wfrag = (unsigned short*)(vxv + BN * 9);
    float*          out   = (float*)d_out;

    int tot1 = (int)(BN * 9);
    feast_pre<<<(tot1 + 255) / 256, 256, 0, stream>>>(x, u, v, cvec, uxc, vxv, (int)BN);
    feast_packW<<<(36864 + 255) / 256, 256, 0, stream>>>(W, wfrag);

    int tiles = (N + 15) / 16;
    feast_main<<<B * tiles, 32, 0, stream>>>(x, adj, bias, uxc, vxv, wfrag, out, N);
}
